// SCNAttnDecoder_54649163874972
// MI455X (gfx1250) — matrix operand order repaired and checked by reading, not compiled, not run on hardware
//
#include <hip/hip_runtime.h>
#include <hip/hip_bf16.h>

// ---------------------------------------------------------------------------
// SCN-LSTM decoder for MI455X (gfx1250, wave32, WMMA).
//   B=128, T=20, E=512, K=512, H=1024, NT=300, V=20000
//   - bf16 WMMA (v_wmma_f32_16x16x32_bf16) for every GEMM, f32 accumulate.
//   - x-side gate GEMMs batched over all timesteps (no h dependence).
//   - 20 short dependent launches for the h-recurrence.
//   - logits = one big batched GEMM [2560,1024]x[1024,20000] after the scan.
//   - double-buffered LDS staging via GLOBAL_LOAD_ASYNC_TO_LDS_B128
//     (ASYNCcnt path), register-staged fallback otherwise.
// ---------------------------------------------------------------------------

#define BB   128
#define TT   20
#define EE   512
#define KK   512
#define HH   1024
#define NTAG 300
#define VV   20000
#define MM   (TT * BB)   // 2560

#if defined(__HIP_DEVICE_COMPILE__) && \
    __has_builtin(__builtin_amdgcn_global_load_async_to_lds_b128)
#define HAVE_ASYNC_LDS 1
#else
#define HAVE_ASYNC_LDS 0
#endif

typedef __attribute__((ext_vector_type(16))) __bf16 v16bf;
typedef __attribute__((ext_vector_type(8)))  float  v8f;
typedef int v4i_ __attribute__((vector_size(16)));   // matches builtin 'V4i'

__device__ __forceinline__ __bf16 f2bf(float f) {
  union { float f; unsigned u; } in; in.f = f;
  unsigned u = in.u;
  u += 0x7FFFu + ((u >> 16) & 1u);           // round-to-nearest-even
  union { unsigned short s; __bf16 b; } out; out.s = (unsigned short)(u >> 16);
  return out.b;
}
__device__ __forceinline__ float bf2f(__bf16 b) {
  union { __bf16 b; unsigned short s; } in; in.b = b;
  union { unsigned u; float f; } out; out.u = ((unsigned)in.s) << 16;
  return out.f;
}
__device__ __forceinline__ float sigmoidf(float x) {
  return 1.0f / (1.0f + __expf(-x));
}

#if HAVE_ASYNC_LDS
__device__ __forceinline__ void async_cp_b128(const __bf16* gsrc, __bf16* ldst) {
  __builtin_amdgcn_global_load_async_to_lds_b128(
      (__attribute__((address_space(1))) v4i_*)gsrc,
      (__attribute__((address_space(3))) v4i_*)ldst, 0, 0);
}
__device__ __forceinline__ void wait_async0() {
#if __has_builtin(__builtin_amdgcn_s_wait_asynccnt)
  __builtin_amdgcn_s_wait_asynccnt(0);
#else
  asm volatile("s_wait_asynccnt 0" ::: "memory");
#endif
}
#endif

// ---------------------------------------------------------------------------
// Tiled transpose + f32->bf16 convert: src f32 [R][C]  ->  dst bf16 [C][R].
// grid.z = batch (per-gate), block (32,8).
// ---------------------------------------------------------------------------
__global__ void k_transpose_bf16(const float* __restrict__ src,
                                 __bf16* __restrict__ dst, int R, int C) {
  __shared__ float tile[32][33];
  size_t bofs = (size_t)blockIdx.z * (size_t)R * (size_t)C;
  src += bofs; dst += bofs;
  int c0 = blockIdx.x * 32, r0 = blockIdx.y * 32;
  for (int i = threadIdx.y; i < 32; i += 8) {
    int r = r0 + i, c = c0 + threadIdx.x;
    tile[i][threadIdx.x] = (r < R && c < C) ? src[(size_t)r * C + c] : 0.0f;
  }
  __syncthreads();
  for (int i = threadIdx.y; i < 32; i += 8) {
    int c = c0 + i, r = r0 + threadIdx.x;
    if (c < C && r < R) dst[(size_t)c * R + r] = f2bf(tile[threadIdx.x][i]);
  }
}

// ---------------------------------------------------------------------------
// Tag-side factors: s2[g][b][k] = sum_t s_tags[b][t]*Wb[g][t][k]  (z in 0..3)
//                   s5[g][b][k] = sum_t s_tags[b][t]*Ub[g][t][k]  (z in 4..7)
// ---------------------------------------------------------------------------
__global__ void k_tags(const float* __restrict__ s_tags,
                       const float* __restrict__ Wb,
                       const float* __restrict__ Ub,
                       float* __restrict__ s2, float* __restrict__ s5) {
  int z = blockIdx.z; int g = z & 3; bool hside = (z >= 4);
  int bk = blockIdx.x * 256 + threadIdx.x;      // b*KK + k
  if (bk >= BB * KK) return;
  int b = bk >> 9;
  int k = bk & (KK - 1);
  const float* W  = (hside ? Ub : Wb) + (size_t)g * NTAG * KK + k;
  const float* st = s_tags + (size_t)b * NTAG;
  float acc = 0.0f;
  for (int t = 0; t < NTAG; ++t) acc += st[t] * W[(size_t)t * KK];
  (hside ? s5 : s2)[(size_t)g * (BB * KK) + bk] = acc;
}

// ---------------------------------------------------------------------------
// Teacher-forcing inputs: xs[(t*B+b)][e] = (t==0) ? 0 : emb[cap[b][t-1]][e]
// ---------------------------------------------------------------------------
__global__ void k_embed(const int* __restrict__ captions,
                        const float* __restrict__ emb,
                        __bf16* __restrict__ xs) {
  long long idx = (long long)blockIdx.x * 256 + threadIdx.x;
  if (idx >= (long long)MM * EE) return;
  int e = (int)(idx & (EE - 1));
  int m = (int)(idx >> 9);
  int t = m / BB, b = m % BB;
  float v = 0.0f;
  if (t > 0) {
    int tok = captions[b * TT + (t - 1)];
    v = emb[(size_t)tok * EE + e];
  }
  xs[idx] = f2bf(v);
}

// h_cur=h0, c_cur=c0, h_bf=bf16(h0)
__global__ void k_init(const float* __restrict__ h0, const float* __restrict__ c0,
                       float* __restrict__ h_cur, float* __restrict__ c_cur,
                       __bf16* __restrict__ h_bf) {
  int idx = blockIdx.x * 256 + threadIdx.x;
  if (idx >= BB * HH) return;
  float h = h0[idx];
  h_cur[idx] = h; c_cur[idx] = c0[idx]; h_bf[idx] = f2bf(h);
}

// ---------------------------------------------------------------------------
// WMMA GEMM core:  C[M,N] = A[M,K] * Bt[N,K]^T   (both operands bf16, K-major)
// 128x128 tile, BK=32, 256 threads = 8 waves, each wave owns 2x4 16x16 tiles.
// Double-buffered LDS; async global->LDS when available.
// Epilogues (template EPI):
//   0: bf16 store, scaled by aux[(m%128)*N + n]          (x1*s2 / h6*s5)
//   1: bf16 store                                        (gx_all)
//   2: f32 store                                         (gh_all)
//   3: f32 store + bias aux[n], scatter m=t*B+b -> [b][t][n]  (logits)
// ---------------------------------------------------------------------------
template <int EPI>
__global__ __launch_bounds__(256) void k_gemm(
    const __bf16* __restrict__ A, const __bf16* __restrict__ Bt,
    void* __restrict__ Cout, const float* __restrict__ aux,
    int M, int N, int K,
    long long aStrideG, long long bStrideG, long long cStrideG,
    long long auxStrideG) {
  __shared__ __attribute__((aligned(16))) __bf16 ldsA[2][128 * 32];
  __shared__ __attribute__((aligned(16))) __bf16 ldsB[2][128 * 32];

  int g = blockIdx.z;
  A  += (size_t)g * aStrideG;
  Bt += (size_t)g * bStrideG;
  if (aux) aux += (size_t)g * auxStrideG;

  int tid  = threadIdx.x;
  int lane = tid & 31, wave = tid >> 5;
  int wm = wave & 3;          // 4 waves along M (32 rows each)
  int wn = wave >> 2;         // 2 waves along N (64 cols each)
  int mBase = blockIdx.y * 128, nBase = blockIdx.x * 128;

  int r  = lane & 15;         // row within 16 (A: M, B: N)
  int kh = lane >> 4;         // K-half selector per ISA VGPR striping

  // staging assignment: each thread copies 2 x 16B for A and for B.
  int sRow = tid >> 2;                    // 0..63 (second chunk: +64)
  int sCol = (tid & 3) << 3;              // bf16 column within 32-wide K tile
  int aR0 = mBase + sRow, aR1 = aR0 + 64; // A rows always in range
  int bR0 = nBase + sRow, bR1 = bR0 + 64; // clamp B rows (edge tile of logits):
  if (bR0 > N - 1) bR0 = N - 1;           // garbage cols are never stored
  if (bR1 > N - 1) bR1 = N - 1;
  int lOfs0 = sRow * 32 + sCol, lOfs1 = lOfs0 + 64 * 32;

  v8f acc[2][4];
#pragma unroll
  for (int i = 0; i < 2; ++i)
#pragma unroll
    for (int j = 0; j < 4; ++j) acc[i][j] = v8f{};

  // ---- prologue: stage K-tile 0 into buffer 0 -----------------------------
#if HAVE_ASYNC_LDS
  async_cp_b128(&A[(size_t)aR0 * K + sCol],  &ldsA[0][lOfs0]);
  async_cp_b128(&A[(size_t)aR1 * K + sCol],  &ldsA[0][lOfs1]);
  async_cp_b128(&Bt[(size_t)bR0 * K + sCol], &ldsB[0][lOfs0]);
  async_cp_b128(&Bt[(size_t)bR1 * K + sCol], &ldsB[0][lOfs1]);
  wait_async0();
#else
  {
    uint4 a0 = *(const uint4*)&A[(size_t)aR0 * K + sCol];
    uint4 a1 = *(const uint4*)&A[(size_t)aR1 * K + sCol];
    uint4 b0 = *(const uint4*)&Bt[(size_t)bR0 * K + sCol];
    uint4 b1 = *(const uint4*)&Bt[(size_t)bR1 * K + sCol];
    *(uint4*)&ldsA[0][lOfs0] = a0;
    *(uint4*)&ldsA[0][lOfs1] = a1;
    *(uint4*)&ldsB[0][lOfs0] = b0;
    *(uint4*)&ldsB[0][lOfs1] = b1;
  }
#endif
  __syncthreads();

  int parity = 0;
  for (int k0 = 0; k0 < K; k0 += 32) {
    bool more = (k0 + 32 < K);
    int kn = k0 + 32;

    // ---- kick off staging of the next K-tile into the other buffer -------
#if HAVE_ASYNC_LDS
    if (more) {
      async_cp_b128(&A[(size_t)aR0 * K + kn + sCol],  &ldsA[parity ^ 1][lOfs0]);
      async_cp_b128(&A[(size_t)aR1 * K + kn + sCol],  &ldsA[parity ^ 1][lOfs1]);
      async_cp_b128(&Bt[(size_t)bR0 * K + kn + sCol], &ldsB[parity ^ 1][lOfs0]);
      async_cp_b128(&Bt[(size_t)bR1 * K + kn + sCol], &ldsB[parity ^ 1][lOfs1]);
    }
#else
    uint4 na0, na1, nb0, nb1;
    if (more) {
      na0 = *(const uint4*)&A[(size_t)aR0 * K + kn + sCol];
      na1 = *(const uint4*)&A[(size_t)aR1 * K + kn + sCol];
      nb0 = *(const uint4*)&Bt[(size_t)bR0 * K + kn + sCol];
      nb1 = *(const uint4*)&Bt[(size_t)bR1 * K + kn + sCol];
    }
    if (kn + 32 < K) {   // speculative prefetch two tiles ahead
      __builtin_prefetch(&A[(size_t)aR0 * K + kn + 32 + sCol], 0, 1);
      __builtin_prefetch(&Bt[(size_t)bR0 * K + kn + 32 + sCol], 0, 1);
    }
#endif

    // ---- fragments per ISA 16-bit A/B VGPR layouts, then 8 WMMAs ----------
    const __bf16* lA = ldsA[parity];
    const __bf16* lB = ldsB[parity];
    v16bf afrag[2], bfrag[4];
#pragma unroll
    for (int ti = 0; ti < 2; ++ti) {
      const __bf16* p = &lA[(wm * 32 + ti * 16 + r) * 32 + kh * 8];
      ((uint4*)&afrag[ti])[0] = *(const uint4*)p;         // K = kh*8    .. +7
      ((uint4*)&afrag[ti])[1] = *(const uint4*)(p + 16);  // K = 16+kh*8 .. +7
    }
#pragma unroll
    for (int tj = 0; tj < 4; ++tj) {
      const __bf16* p = &lB[(wn * 64 + tj * 16 + r) * 32 + kh * 8];
      ((uint4*)&bfrag[tj])[0] = *(const uint4*)p;         // K = kh*8    .. +7
      ((uint4*)&bfrag[tj])[1] = *(const uint4*)(p + 16);  // K = 16+kh*8 .. +7
    }
#pragma unroll
    for (int ti = 0; ti < 2; ++ti)
#pragma unroll
      for (int tj = 0; tj < 4; ++tj)
        acc[ti][tj] = __builtin_amdgcn_wmma_f32_16x16x32_bf16(
            false, afrag[ti], false, bfrag[tj], (short)0, acc[ti][tj],
            false, false);

    // ---- complete next-tile staging, flip buffers -------------------------
#if HAVE_ASYNC_LDS
    if (more) wait_async0();
#else
    if (more) {
      *(uint4*)&ldsA[parity ^ 1][lOfs0] = na0;
      *(uint4*)&ldsA[parity ^ 1][lOfs1] = na1;
      *(uint4*)&ldsB[parity ^ 1][lOfs0] = nb0;
      *(uint4*)&ldsB[parity ^ 1][lOfs1] = nb1;
    }
#endif
    __syncthreads();
    parity ^= 1;
  }

  // ---- epilogue: C/D layout lane(r)=N, vgpr e -> M = kh*8+e ---------------
#pragma unroll
  for (int ti = 0; ti < 2; ++ti)
#pragma unroll
    for (int tj = 0; tj < 4; ++tj)
#pragma unroll
      for (int e = 0; e < 8; ++e) {
        int gm = mBase + wm * 32 + ti * 16 + kh * 8 + e;
        int gn = nBase + wn * 64 + tj * 16 + r;
        if (gn >= N) continue;
        float v = acc[ti][tj][e];
        if (EPI == 0) {
          float s = aux[(size_t)(gm & (BB - 1)) * N + gn];
          ((__bf16*)Cout)[(size_t)g * cStrideG + (size_t)gm * N + gn] =
              f2bf(v * s);
        } else if (EPI == 1) {
          ((__bf16*)Cout)[(size_t)g * cStrideG + (size_t)gm * N + gn] = f2bf(v);
        } else if (EPI == 2) {
          ((float*)Cout)[(size_t)g * cStrideG + (size_t)gm * N + gn] = v;
        } else {
          int t = gm / BB, b = gm % BB;   // m = t*B + b  ->  out[b][t][n]
          ((float*)Cout)[(size_t)b * TT * VV + (size_t)t * VV + gn] =
              v + aux[gn];
        }
      }
}

// ---------------------------------------------------------------------------
// Fused LSTM cell for step t: gates = gh_all[g] + gx_all[g][t] + bias[g]
// ---------------------------------------------------------------------------
__global__ void k_cell(const float* __restrict__ gh_all,
                       const __bf16* __restrict__ gx_all,
                       const float* __restrict__ bias,
                       float* __restrict__ c_cur, float* __restrict__ h_cur,
                       __bf16* __restrict__ h_bf, __bf16* __restrict__ Hs,
                       int t) {
  int idx = blockIdx.x * 256 + threadIdx.x;   // b*HH + h
  if (idx >= BB * HH) return;
  int b = idx >> 10, h = idx & (HH - 1);
  size_t m  = (size_t)t * BB + b;
  size_t gS = (size_t)MM * HH;                // gx gate stride
  float gv[4];
#pragma unroll
  for (int g = 0; g < 4; ++g)
    gv[g] = gh_all[(size_t)g * BB * HH + idx] +
            bf2f(gx_all[(size_t)g * gS + m * HH + h]) + bias[g * HH + h];
  float i_ = sigmoidf(gv[0]);
  float f_ = sigmoidf(gv[1]);
  float o_ = sigmoidf(gv[2]);
  float cg = tanhf(gv[3]);
  float cn = f_ * c_cur[idx] + i_ * cg;
  float hn = o_ * tanhf(cn);
  c_cur[idx] = cn;
  h_cur[idx] = hn;
  __bf16 hb = f2bf(hn);
  h_bf[idx] = hb;
  Hs[m * HH + h] = hb;
}

// ---------------------------------------------------------------------------
extern "C" void kernel_launch(void* const* d_in, const int* in_sizes, int n_in,
                              void* d_out, int out_size, void* d_ws,
                              size_t ws_size, hipStream_t stream) {
  const float* s_tags   = (const float*)d_in[0];
  const int*   captions = (const int*)d_in[1];
  const float* h0   = (const float*)d_in[2];
  const float* c0   = (const float*)d_in[3];
  const float* Wa   = (const float*)d_in[4];
  const float* Wb   = (const float*)d_in[5];
  const float* Wc   = (const float*)d_in[6];
  const float* Ua   = (const float*)d_in[7];
  const float* Ub   = (const float*)d_in[8];
  const float* Uc   = (const float*)d_in[9];
  const float* bgat = (const float*)d_in[10];
  const float* emb  = (const float*)d_in[11];
  const float* outW = (const float*)d_in[12];
  const float* outb = (const float*)d_in[13];
  float* out = (float*)d_out;

  char* ws = (char*)d_ws;
  size_t off = 0;
  auto alloc = [&](size_t bytes) -> void* {
    void* p = ws + off;
    off = (off + bytes + 255) & ~(size_t)255;
    return p;
  };

  __bf16* WaT    = (__bf16*)alloc((size_t)4 * KK * EE * 2);     // [4][K][E]
  __bf16* UaT    = (__bf16*)alloc((size_t)4 * KK * HH * 2);     // [4][K][H]
  __bf16* WcT    = (__bf16*)alloc((size_t)4 * HH * KK * 2);     // [4][H][K]
  __bf16* UcT    = (__bf16*)alloc((size_t)4 * HH * KK * 2);     // [4][H][K]
  __bf16* outWt  = (__bf16*)alloc((size_t)VV * HH * 2);         // [V][H]
  float*  s2     = (float*) alloc((size_t)4 * BB * KK * 4);
  float*  s5     = (float*) alloc((size_t)4 * BB * KK * 4);
  __bf16* xs     = (__bf16*)alloc((size_t)MM * EE * 2);
  __bf16* a1     = (__bf16*)alloc((size_t)4 * MM * KK * 2);     // (x1*s2)
  __bf16* gx_all = (__bf16*)alloc((size_t)4 * MM * HH * 2);
  __bf16* a5     = (__bf16*)alloc((size_t)4 * BB * KK * 2);     // (h6*s5)
  float*  gh_all = (float*) alloc((size_t)4 * BB * HH * 4);
  float*  h_cur  = (float*) alloc((size_t)BB * HH * 4);
  float*  c_cur  = (float*) alloc((size_t)BB * HH * 4);
  __bf16* h_bf   = (__bf16*)alloc((size_t)BB * HH * 2);
  __bf16* Hs     = (__bf16*)alloc((size_t)MM * HH * 2);

  dim3 tb(32, 8);
  // weight transposes (f32 -> bf16, [R][C] -> [C][R]); grid.z = gate
  k_transpose_bf16<<<dim3(KK / 32, EE / 32, 4), tb, 0, stream>>>(Wa, WaT, EE, KK);
  k_transpose_bf16<<<dim3(KK / 32, HH / 32, 4), tb, 0, stream>>>(Ua, UaT, HH, KK);
  k_transpose_bf16<<<dim3(HH / 32, KK / 32, 4), tb, 0, stream>>>(Wc, WcT, KK, HH);
  k_transpose_bf16<<<dim3(HH / 32, KK / 32, 4), tb, 0, stream>>>(Uc, UcT, KK, HH);
  k_transpose_bf16<<<dim3(VV / 32, HH / 32, 1), tb, 0, stream>>>(outW, outWt, HH, VV);

  k_tags<<<dim3((BB * KK + 255) / 256, 1, 8), 256, 0, stream>>>(
      s_tags, Wb, Ub, s2, s5);
  k_embed<<<((long long)MM * EE + 255) / 256, 256, 0, stream>>>(captions, emb, xs);
  k_init<<<(BB * HH + 255) / 256, 256, 0, stream>>>(h0, c0, h_cur, c_cur, h_bf);

  // x-side for all timesteps: a1 = bf16((xs @ Wa[g]) * s2[g]),  [4][2560][512]
  k_gemm<0><<<dim3(KK / 128, MM / 128, 4), 256, 0, stream>>>(
      xs, WaT, a1, s2, MM, KK, EE,
      0LL, (long long)KK * EE, (long long)MM * KK, (long long)BB * KK);
  // gx_all = bf16(a1 @ Wc[g]),  [4][2560][1024]
  k_gemm<1><<<dim3(HH / 128, MM / 128, 4), 256, 0, stream>>>(
      a1, WcT, gx_all, nullptr, MM, HH, KK,
      (long long)MM * KK, (long long)HH * KK, (long long)MM * HH, 0LL);

  // sequential recurrence
  for (int t = 0; t < TT; ++t) {
    // a5 = bf16((h @ Ua[g]) * s5[g]),  [4][128][512]
    k_gemm<0><<<dim3(KK / 128, BB / 128, 4), 256, 0, stream>>>(
        h_bf, UaT, a5, s5, BB, KK, HH,
        0LL, (long long)KK * HH, (long long)BB * KK, (long long)BB * KK);
    // gh_all = f32(a5 @ Uc[g]),  [4][128][1024]
    k_gemm<2><<<dim3(HH / 128, BB / 128, 4), 256, 0, stream>>>(
        a5, UcT, gh_all, nullptr, BB, HH, KK,
        (long long)BB * KK, (long long)HH * KK, (long long)BB * HH, 0LL);
    k_cell<<<(BB * HH + 255) / 256, 256, 0, stream>>>(
        gh_all, gx_all, bgat, c_cur, h_cur, h_bf, Hs, t);
  }

  // logits: out[b][t][v] = Hs[t*B+b] @ out_W + out_b   (105 GFLOP, one GEMM)
  k_gemm<3><<<dim3((VV + 127) / 128, MM / 128, 1), 256, 0, stream>>>(
      Hs, outWt, out, outb, MM, VV, HH,
      0LL, 0LL, 0LL, 0LL);
  (void)in_sizes; (void)n_in; (void)out_size; (void)ws_size;
}
